// PipelinedMoEBlock_66133906424316
// MI455X (gfx1250) — matrix operand order repaired and checked by reading, not compiled, not run on hardware
//
#include <hip/hip_runtime.h>
#include <hip/hip_bf16.h>

typedef __attribute__((ext_vector_type(16))) _Float16 v16h;
typedef __attribute__((ext_vector_type(8)))  float    v8f;

#define BATCH  4
#define SEQ    512
#define DMODEL 768
#define NHEAD  12
#define DHEAD  64
#define NEXP   8
#define FFDIM  2048
#define QKV_LD (3 * DMODEL)
#define NTOK   (BATCH * SEQ)

// ---------------- WMMA fragment loaders (ISA 7.12.2 layouts, wave32) ----------------

// A-matrix 16x32 f16 (row-major [M x K]): lane l holds row M = l&15; halves 0..7 =
// K = k0+8*(l>>4)+j, halves 8..15 = K = k0+16+8*(l>>4)+j. Two 16B runs -> b128 loads.
__device__ __forceinline__ v16h load_a_frag(const _Float16* __restrict__ A, int lda,
                                            int row0, int k0, int lane) {
  const int m  = row0 + (lane & 15);
  const int kb = k0 + ((lane >> 4) << 3);
  const _Float16* p0 = A + (size_t)m * lda + kb;
  const _Float16* p1 = p0 + 16;
  v16h r;
#pragma unroll
  for (int i = 0; i < 8; ++i) { r[i] = p0[i]; r[8 + i] = p1[i]; }
  return r;
}

// B-fragment from a transposed operand BT stored row-major [N x K]:
// lane l holds col N = n0+(l&15); element e<8 = BT[n][k0 + 8*(l>>4) + e], e>=8 = BT[n][k0 + 16 + 8*(l>>4) + e-8].
// Two 16B runs -> two b128 loads (works for global or LDS pointers).
__device__ __forceinline__ v16h load_bT_frag(const _Float16* __restrict__ X, int ldx,
                                             int n0, int k0, int lane) {
  const int n  = n0 + (lane & 15);
  const int kb = k0 + ((lane >> 4) << 3);
  const _Float16* p = X + (size_t)n * ldx + kb;
  v16h r;
#pragma unroll
  for (int i = 0; i < 8; ++i) { r[i] = p[i]; r[8 + i] = p[16 + i]; }
  return r;
}

__device__ __forceinline__ float gelu_tanh(float x) {
  const float c = 0.7978845608028654f;
  return 0.5f * x * (1.0f + tanhf(c * (x + 0.044715f * x * x * x)));
}

// ---------------- conversion / repack kernels ----------------

// s: [batch][rows x cols] f32 row-major  ->  d: [batch][cols x rows] f16 row-major
__global__ void transpose_f32_to_f16_kernel(const float* __restrict__ s,
                                            _Float16* __restrict__ d,
                                            int rows, int cols) {
  const size_t total = (size_t)rows * cols;
  s += (size_t)blockIdx.y * total;
  d += (size_t)blockIdx.y * total;
  for (size_t i = (size_t)blockIdx.x * blockDim.x + threadIdx.x; i < total;
       i += (size_t)gridDim.x * blockDim.x) {
    const int r = (int)(i / cols), c = (int)(i % cols);
    d[(size_t)c * rows + r] = (_Float16)s[i];
  }
}

__global__ void copy_f32_kernel(const float* __restrict__ s, float* __restrict__ d, long n) {
  long i = (long)blockIdx.x * blockDim.x + threadIdx.x;
  const long st = (long)gridDim.x * blockDim.x;
  for (; i < n; i += st) d[i] = s[i];
}

// vt[b][h][d][s] = qkv16[b,s, 2*DMODEL + h*DHEAD + d]   (V in per-head [Dh x S] layout)
__global__ void repack_v_kernel(const _Float16* __restrict__ qkv16, _Float16* __restrict__ vt) {
  const int total = BATCH * NHEAD * DHEAD * SEQ;
  for (int idx = blockIdx.x * 256 + threadIdx.x; idx < total; idx += gridDim.x * 256) {
    const int s = idx & (SEQ - 1);
    const int d = (idx >> 9) & (DHEAD - 1);
    const int h = (idx >> 15) % NHEAD;
    const int b = idx / (SEQ * DHEAD * NHEAD);
    vt[idx] = qkv16[((size_t)b * SEQ + s) * QKV_LD + 2 * DMODEL + h * DHEAD + d];
  }
}

// ---------------- LayerNorm (one 256-thread block per token) ----------------

__global__ __launch_bounds__(256) void ln_kernel(const float* __restrict__ x,
                                                 const float* __restrict__ g,
                                                 const float* __restrict__ b,
                                                 _Float16* __restrict__ out16) {
  __shared__ float red[256];
  const int tok = blockIdx.x;
  const float* xr = x + (size_t)tok * DMODEL;
  float s = 0.f, s2 = 0.f;
  for (int d = threadIdx.x; d < DMODEL; d += 256) { float v = xr[d]; s += v; s2 += v * v; }
  red[threadIdx.x] = s; __syncthreads();
  for (int st = 128; st > 0; st >>= 1) { if (threadIdx.x < st) red[threadIdx.x] += red[threadIdx.x + st]; __syncthreads(); }
  const float mean = red[0] / DMODEL; __syncthreads();
  red[threadIdx.x] = s2; __syncthreads();
  for (int st = 128; st > 0; st >>= 1) { if (threadIdx.x < st) red[threadIdx.x] += red[threadIdx.x + st]; __syncthreads(); }
  const float var = red[0] / DMODEL - mean * mean;
  const float rstd = rsqrtf(var + 1e-5f);
  for (int d = threadIdx.x; d < DMODEL; d += 256)
    out16[(size_t)tok * DMODEL + d] = (_Float16)((xr[d] - mean) * rstd * g[d] + b[d]);
}

// ---------------- LN2 + router softmax + top-2 renormalized combine weights ----------------

__global__ __launch_bounds__(256) void ln2_router_kernel(const float* __restrict__ resid,
                                                         const float* __restrict__ g,
                                                         const float* __restrict__ b,
                                                         const float* __restrict__ Wr,
                                                         _Float16* __restrict__ m16,
                                                         float* __restrict__ we) {
  __shared__ float red[256];
  __shared__ float sacc[NEXP];
  const int tok = blockIdx.x;
  const float* xr = resid + (size_t)tok * DMODEL;
  float s = 0.f, s2 = 0.f;
  for (int d = threadIdx.x; d < DMODEL; d += 256) { float v = xr[d]; s += v; s2 += v * v; }
  red[threadIdx.x] = s; __syncthreads();
  for (int st = 128; st > 0; st >>= 1) { if (threadIdx.x < st) red[threadIdx.x] += red[threadIdx.x + st]; __syncthreads(); }
  const float mean = red[0] / DMODEL; __syncthreads();
  red[threadIdx.x] = s2; __syncthreads();
  for (int st = 128; st > 0; st >>= 1) { if (threadIdx.x < st) red[threadIdx.x] += red[threadIdx.x + st]; __syncthreads(); }
  const float var = red[0] / DMODEL - mean * mean;
  const float rstd = rsqrtf(var + 1e-5f);
  if (threadIdx.x < NEXP) sacc[threadIdx.x] = 0.f;
  __syncthreads();
  float p[NEXP];
#pragma unroll
  for (int e = 0; e < NEXP; ++e) p[e] = 0.f;
  for (int d = threadIdx.x; d < DMODEL; d += 256) {
    const float mv = (xr[d] - mean) * rstd * g[d] + b[d];
    m16[(size_t)tok * DMODEL + d] = (_Float16)mv;
#pragma unroll
    for (int e = 0; e < NEXP; ++e) p[e] += mv * Wr[(size_t)d * NEXP + e];
  }
#pragma unroll
  for (int e = 0; e < NEXP; ++e) atomicAdd(&sacc[e], p[e]);
  __syncthreads();
  if (threadIdx.x == 0) {
    float mx = sacc[0];
#pragma unroll
    for (int e = 1; e < NEXP; ++e) mx = fmaxf(mx, sacc[e]);
    float pr[NEXP], se = 0.f;
#pragma unroll
    for (int e = 0; e < NEXP; ++e) { pr[e] = __expf(sacc[e] - mx); se += pr[e]; }
#pragma unroll
    for (int e = 0; e < NEXP; ++e) pr[e] /= se;
    int i1 = 0;
#pragma unroll
    for (int e = 1; e < NEXP; ++e) if (pr[e] > pr[i1]) i1 = e;
    int i2 = (i1 == 0) ? 1 : 0;
#pragma unroll
    for (int e = 0; e < NEXP; ++e) if (e != i1 && pr[e] > pr[i2]) i2 = e;
    const float wsum = pr[i1] + pr[i2];
#pragma unroll
    for (int e = 0; e < NEXP; ++e) we[(size_t)tok * NEXP + e] = 0.f;
    we[(size_t)tok * NEXP + i1] = pr[i1] / wsum;
    we[(size_t)tok * NEXP + i2] = pr[i2] / wsum;
  }
}

// ---------------- WMMA GEMM with async-to-LDS double-buffered B panel ----------------
// Block = 4 waves. Block tile: 128(M) x 64(N). Wave tile: 32(M) x 64(N) -> 8 WMMA / K-step.
// BT is the transposed operand [N x Kd] row-major; a 64x32 panel is staged into LDS with
// global_load_async_to_lds_b128, double-buffered and fenced with s_wait_asynccnt.
// All 4 B-fragments are preloaded into distinct registers before the WMMA burst so the
// ds_loads issue as one clause and the 8 WMMAs run without interleaved dscnt waits.

enum { MODE_OUT16 = 0, MODE_GELU16 = 1, MODE_ADD32 = 2, MODE_SCALEACC32 = 3 };

#define BP_LD 40  // panel row stride in halves: 80B, keeps b128 alignment, staggers banks

__global__ __launch_bounds__(128)
void wmma_gemm_kernel(const _Float16* __restrict__ A, int lda,
                      const _Float16* __restrict__ BT, int ldbt,
                      const float* __restrict__ bias,
                      const float* __restrict__ addsrc,
                      const float* __restrict__ rowscale, int rs_stride,
                      _Float16* __restrict__ out16, float* __restrict__ out32, int ldo,
                      int N, int Kd, int mode) {
  __shared__ _Float16 bpanel[2][64][BP_LD];
  const int lane = threadIdx.x;
  const int wid  = threadIdx.y;
  const int tid  = wid * 32 + lane;
  const int n0 = blockIdx.x * 64;
  const int m0 = blockIdx.y * 128 + wid * 32;
  if (n0 >= N) return;

  // stage one 64x32 B panel: 128 lanes x 2 async b128 copies (16B per lane per op)
  auto stage = [&](int buf, int k0) {
    int chunk = tid;
#pragma unroll
    for (int rep = 0; rep < 2; ++rep, chunk += 128) {
      const int nrow = chunk >> 2;          // 0..63
      const int kc   = (chunk & 3) * 8;     // 0,8,16,24 halves
      const _Float16* gp = BT + (size_t)(n0 + nrow) * ldbt + k0 + kc;
      const unsigned loff = (unsigned)(uintptr_t)&bpanel[buf][nrow][kc];
      const unsigned long long ga = (unsigned long long)(uintptr_t)gp;
      asm volatile("global_load_async_to_lds_b128 %0, %1, off"
                   :: "v"(loff), "v"(ga) : "memory");
    }
  };

  v8f acc[2][4] = {};
  const int nk = Kd / 32;
  stage(0, 0);
  for (int it = 0; it < nk; ++it) {
    const int buf = it & 1;
    if (it + 1 < nk) {
      stage(buf ^ 1, (it + 1) * 32);                      // prefetch next panel
      asm volatile("s_wait_asynccnt 0x2" ::: "memory");   // current panel landed
    } else {
      asm volatile("s_wait_asynccnt 0x0" ::: "memory");
    }
    __syncthreads();

    const int k0 = it * 32;
    const v16h a0 = load_a_frag(A, lda, m0, k0, lane);
    const v16h a1 = load_a_frag(A, lda, m0 + 16, k0, lane);
    v16h bf[4];
#pragma unroll
    for (int j = 0; j < 4; ++j)
      bf[j] = load_bT_frag(&bpanel[buf][0][0], BP_LD, j * 16, 0, lane);
#pragma unroll
    for (int j = 0; j < 4; ++j) {
      acc[0][j] = __builtin_amdgcn_wmma_f32_16x16x32_f16(false, a0, false, bf[j], (short)0, acc[0][j], false, false);
      acc[1][j] = __builtin_amdgcn_wmma_f32_16x16x32_f16(false, a1, false, bf[j], (short)0, acc[1][j], false, false);
    }
    __syncthreads();
  }

  const int nlo = lane & 15;
  const int g8  = 8 * (lane >> 4);
#pragma unroll
  for (int ms = 0; ms < 2; ++ms) {
#pragma unroll
    for (int j = 0; j < 4; ++j) {
      const int n = n0 + j * 16 + nlo;
      const float bv = bias ? bias[n] : 0.f;
#pragma unroll
      for (int i = 0; i < 8; ++i) {
        const int m = m0 + ms * 16 + g8 + i;
        float v = acc[ms][j][i] + bv;
        const size_t idx = (size_t)m * ldo + n;
        if (mode == MODE_OUT16)        out16[idx] = (_Float16)v;
        else if (mode == MODE_GELU16)  out16[idx] = (_Float16)gelu_tanh(v);
        else if (mode == MODE_ADD32)   out32[idx] = addsrc[idx] + v;
        else                           out32[idx] += rowscale[(size_t)m * rs_stride] * v;
      }
    }
  }
}

// ---------------- fused causal flash attention: one wave per 16-query tile ----------------

__global__ __launch_bounds__(32)
void flash_attn_kernel(const _Float16* __restrict__ qkv16,
                       const _Float16* __restrict__ vt,
                       _Float16* __restrict__ o16) {
  __shared__ _Float16 lds_p[16][34];  // P tile staging (C-frag layout -> A-frag layout)
  const int lane = threadIdx.x;
  const int qt = blockIdx.x;
  const int h  = blockIdx.y;
  const int b  = blockIdx.z;
  const int q0 = qt * 16;
  const int g   = lane >> 4;
  const int nlo = lane & 15;

  const _Float16* Qb  = qkv16 + (size_t)b * SEQ * QKV_LD + h * DHEAD;
  const _Float16* Kb  = Qb + DMODEL;
  const _Float16* Vtb = vt + (size_t)(b * NHEAD + h) * DHEAD * SEQ;  // [Dh x S]

  v16h aQ0 = load_a_frag(Qb, QKV_LD, q0, 0, lane);
  v16h aQ1 = load_a_frag(Qb, QKV_LD, q0, 32, lane);
  const _Float16 sc = (_Float16)0.125f;  // 1/sqrt(64)
#pragma unroll
  for (int i = 0; i < 16; ++i) { aQ0[i] = aQ0[i] * sc; aQ1[i] = aQ1[i] * sc; }

  v8f oacc[4] = {};
  float mrow[8], lrow[8];
#pragma unroll
  for (int i = 0; i < 8; ++i) { mrow[i] = -1e30f; lrow[i] = 0.f; }

  for (int kt = 0; kt <= qt; kt += 2) {
    float sA[8], sB[8];
    {
      v8f c = {};
      v16h bk = load_bT_frag(Kb, QKV_LD, kt * 16, 0, lane);
      c = __builtin_amdgcn_wmma_f32_16x16x32_f16(false, aQ0, false, bk, (short)0, c, false, false);
      bk = load_bT_frag(Kb, QKV_LD, kt * 16, 32, lane);
      c = __builtin_amdgcn_wmma_f32_16x16x32_f16(false, aQ1, false, bk, (short)0, c, false, false);
#pragma unroll
      for (int i = 0; i < 8; ++i) sA[i] = c[i];
    }
    const bool haveB = (kt + 1) <= qt;
    if (haveB) {
      v8f c = {};
      v16h bk = load_bT_frag(Kb, QKV_LD, (kt + 1) * 16, 0, lane);
      c = __builtin_amdgcn_wmma_f32_16x16x32_f16(false, aQ0, false, bk, (short)0, c, false, false);
      bk = load_bT_frag(Kb, QKV_LD, (kt + 1) * 16, 32, lane);
      c = __builtin_amdgcn_wmma_f32_16x16x32_f16(false, aQ1, false, bk, (short)0, c, false, false);
#pragma unroll
      for (int i = 0; i < 8; ++i) sB[i] = c[i];
    } else {
#pragma unroll
      for (int i = 0; i < 8; ++i) sB[i] = -1e30f;
    }

    const int tokA = kt * 16 + nlo;
    const int tokB = (kt + 1) * 16 + nlo;
    float pA[8], pB[8];
#pragma unroll
    for (int i = 0; i < 8; ++i) {
      const int qrow = q0 + i + 8 * g;
      if (tokA > qrow) sA[i] = -1e30f;
      if (tokB > qrow) sB[i] = -1e30f;
      float rm = fmaxf(sA[i], sB[i]);
#pragma unroll
      for (int off = 8; off > 0; off >>= 1) rm = fmaxf(rm, __shfl_xor(rm, off, 32));
      const float mnew = fmaxf(mrow[i], rm);
      const float alpha = __expf(mrow[i] - mnew);
      mrow[i] = mnew;
      pA[i] = __expf(sA[i] - mnew);
      pB[i] = __expf(sB[i] - mnew);
      float rs = pA[i] + pB[i];
#pragma unroll
      for (int off = 8; off > 0; off >>= 1) rs += __shfl_xor(rs, off, 32);
      lrow[i] = lrow[i] * alpha + rs;
#pragma unroll
      for (int j = 0; j < 4; ++j) oacc[j][i] *= alpha;
    }

    __syncthreads();
#pragma unroll
    for (int i = 0; i < 8; ++i) {
      lds_p[i + 8 * g][nlo]      = (_Float16)pA[i];
      lds_p[i + 8 * g][16 + nlo] = (_Float16)pB[i];
    }
    __syncthreads();
    v16h aP;
    {
      const int kb = g * 8;
#pragma unroll
      for (int j = 0; j < 8; ++j) {
        aP[j]     = lds_p[nlo][kb + j];
        aP[8 + j] = lds_p[nlo][16 + kb + j];
      }
    }
    __syncthreads();

#pragma unroll
    for (int j = 0; j < 4; ++j) {
      const v16h bv = load_bT_frag(Vtb, SEQ, j * 16, kt * 16, lane);
      oacc[j] = __builtin_amdgcn_wmma_f32_16x16x32_f16(false, aP, false, bv, (short)0, oacc[j], false, false);
    }
  }

#pragma unroll
  for (int j = 0; j < 4; ++j) {
#pragma unroll
    for (int i = 0; i < 8; ++i) {
      const int row = q0 + i + 8 * g;
      const int col = h * DHEAD + j * 16 + nlo;
      o16[((size_t)b * SEQ + row) * DMODEL + col] = (_Float16)(oacc[j][i] / lrow[i]);
    }
  }
}

// ---------------- host orchestration ----------------

extern "C" void kernel_launch(void* const* d_in, const int* in_sizes, int n_in,
                              void* d_out, int out_size, void* d_ws, size_t ws_size,
                              hipStream_t stream) {
  (void)in_sizes; (void)n_in; (void)out_size; (void)ws_size;
  const float* x     = (const float*)d_in[0];
  const float* ln1_g = (const float*)d_in[1];
  const float* ln1_b = (const float*)d_in[2];
  const float* Wqkv  = (const float*)d_in[3];
  const float* bqkv  = (const float*)d_in[4];
  const float* Wo    = (const float*)d_in[5];
  const float* bo    = (const float*)d_in[6];
  const float* ln2_g = (const float*)d_in[7];
  const float* ln2_b = (const float*)d_in[8];
  const float* Wr    = (const float*)d_in[9];
  const float* W1    = (const float*)d_in[10];
  const float* b1    = (const float*)d_in[11];
  const float* W2    = (const float*)d_in[12];
  const float* b2    = (const float*)d_in[13];
  float* out = (float*)d_out;

  char* ws = (char*)d_ws;
  size_t off = 0;
  auto alloc = [&](size_t bytes) -> char* {
    char* p = ws + off;
    off = (off + bytes + 255) & ~(size_t)255;
    return p;
  };
  _Float16* h16    = (_Float16*)alloc((size_t)NTOK * DMODEL * 2);
  _Float16* qkv16  = (_Float16*)alloc((size_t)NTOK * QKV_LD * 2);
  _Float16* vt16   = (_Float16*)alloc((size_t)BATCH * NHEAD * DHEAD * SEQ * 2);
  _Float16* o16    = (_Float16*)alloc((size_t)NTOK * DMODEL * 2);
  float*    resid  = (float*)   alloc((size_t)NTOK * DMODEL * 4);
  _Float16* m16    = (_Float16*)alloc((size_t)NTOK * DMODEL * 2);
  float*    we     = (float*)   alloc((size_t)NTOK * NEXP * 4);
  _Float16* hid16  = (_Float16*)alloc((size_t)NTOK * FFDIM * 2);
  _Float16* WqkvT  = (_Float16*)alloc((size_t)QKV_LD * DMODEL * 2);   // [2304 x 768]
  _Float16* WoT    = (_Float16*)alloc((size_t)DMODEL * DMODEL * 2);   // [768 x 768]
  _Float16* W1T    = (_Float16*)alloc((size_t)NEXP * FFDIM * DMODEL * 2);  // [e][2048 x 768]
  _Float16* W2T    = (_Float16*)alloc((size_t)NEXP * DMODEL * FFDIM * 2);  // [e][768 x 2048]

  const dim3 gblk(32, 4, 1);

  // 1) weights f32 -> f16, transposed to [N x K] for contiguous B-fragment loads
  transpose_f32_to_f16_kernel<<<dim3(2048, 1), 256, 0, stream>>>(Wqkv, WqkvT, DMODEL, QKV_LD);
  transpose_f32_to_f16_kernel<<<dim3(1024, 1), 256, 0, stream>>>(Wo, WoT, DMODEL, DMODEL);
  transpose_f32_to_f16_kernel<<<dim3(1024, NEXP), 256, 0, stream>>>(W1, W1T, DMODEL, FFDIM);
  transpose_f32_to_f16_kernel<<<dim3(1024, NEXP), 256, 0, stream>>>(W2, W2T, FFDIM, DMODEL);

  // 2) LN1
  ln_kernel<<<NTOK, 256, 0, stream>>>(x, ln1_g, ln1_b, h16);

  // 3) QKV = h @ Wqkv + bqkv  -> f16 [NTOK, 2304]
  wmma_gemm_kernel<<<dim3(QKV_LD / 64, NTOK / 128), gblk, 0, stream>>>(
      h16, DMODEL, WqkvT, DMODEL, bqkv, nullptr, nullptr, 0,
      qkv16, nullptr, QKV_LD, QKV_LD, DMODEL, MODE_OUT16);

  // 4) repack V per-head transposed, then fused causal flash attention -> o16
  repack_v_kernel<<<4096, 256, 0, stream>>>(qkv16, vt16);
  flash_attn_kernel<<<dim3(SEQ / 16, NHEAD, BATCH), 32, 0, stream>>>(qkv16, vt16, o16);

  // 5) resid = x + o @ Wo + bo   (f32)
  wmma_gemm_kernel<<<dim3(DMODEL / 64, NTOK / 128), gblk, 0, stream>>>(
      o16, DMODEL, WoT, DMODEL, bo, x, nullptr, 0,
      nullptr, resid, DMODEL, DMODEL, DMODEL, MODE_ADD32);

  // 6) LN2 + router softmax + top-2 combine weights
  ln2_router_kernel<<<NTOK, 256, 0, stream>>>(resid, ln2_g, ln2_b, Wr, m16, we);

  // 7) out = resid, then accumulate expert contributions
  copy_f32_kernel<<<4096, 256, 0, stream>>>(resid, out, (long)NTOK * DMODEL);

  // 8) dense expert loop: hid = gelu(m@W1[e]+b1[e]); out += we[:,e] * (hid@W2[e]+b2[e])
  for (int e = 0; e < NEXP; ++e) {
    wmma_gemm_kernel<<<dim3(FFDIM / 64, NTOK / 128), gblk, 0, stream>>>(
        m16, DMODEL, W1T + (size_t)e * FFDIM * DMODEL, DMODEL, b1 + (size_t)e * FFDIM,
        nullptr, nullptr, 0, hid16, nullptr, FFDIM, FFDIM, DMODEL, MODE_GELU16);
    wmma_gemm_kernel<<<dim3(DMODEL / 64, NTOK / 128), gblk, 0, stream>>>(
        hid16, FFDIM, W2T + (size_t)e * DMODEL * FFDIM, FFDIM, b2 + (size_t)e * DMODEL,
        nullptr, we + e, NEXP, nullptr, out, DMODEL, DMODEL, FFDIM, MODE_SCALEACC32);
  }
}
